// TransformerConv_24979529793755
// MI455X (gfx1250) — compile-verified, matrix operand order checked by reading, not run on hardware
//
#include <hip/hip_runtime.h>

// ---- Sizes (fixed by the reference) ----
#define NHEAD   8
#define EMB     512
#define HEAD_D  64
#define SLEN    128
#define BSZ     16
#define ROWS    (SLEN * BSZ)        // 2048 token rows
#define NEG_INF (-1e9f)

typedef __attribute__((ext_vector_type(16))) __bf16 v16bf;
typedef __attribute__((ext_vector_type(8)))  float  v8f;
typedef __attribute__((ext_vector_type(4)))  unsigned int u32x4;
typedef __attribute__((ext_vector_type(8)))  int          i32x8;
typedef __attribute__((ext_vector_type(4)))  int          i32x4;

#if __has_builtin(__builtin_amdgcn_tensor_load_to_lds)
#define HAVE_TDM 1
#else
#define HAVE_TDM 0
#endif

// ---------- bf16 helpers ----------
__device__ __forceinline__ unsigned short f2bf(float f) {
    unsigned u = __builtin_bit_cast(unsigned, f);
    unsigned r = (u + 0x7FFFu + ((u >> 16) & 1u)) >> 16;   // round-to-nearest-even
    return (unsigned short)r;
}
__device__ __forceinline__ float bf2f(unsigned short h) {
    unsigned u = ((unsigned)h) << 16;
    return __builtin_bit_cast(float, u);
}

// Load one 16x32 bf16 WMMA fragment (A layout; B identical with matrix stored
// N-major K-contiguous). p -> element [row0, k0]; per lane: row = lane&15,
// K-elems 0..7 at kb, 8..15 at kb+16, kb = ((lane>>4)<<3).
__device__ __forceinline__ v16bf load_bf16_frag(const unsigned short* p, int ld, int lane) {
    int row = lane & 15;
    int kb  = (lane >> 4) << 3;
    const unsigned short* q = p + row * ld + kb;
    union { v16bf v; uint4 u[2]; } f;
    f.u[0] = *reinterpret_cast<const uint4*>(q);
    f.u[1] = *reinterpret_cast<const uint4*>(q + 16);
    return f.v;
}

#if HAVE_TDM
// Issue one TDM 2D tile load: tile_w x tile_h f32 elements, row stride
// `stride` elements, from global `gaddr` into LDS byte offset `lds_off`.
// Wave-level op (EXEC ignored); call from ONE wave only. Tracked by TENSORcnt.
__device__ __forceinline__ void tdm_load_tile_f32(const void* gaddr, unsigned lds_off,
                                                  unsigned tile_w, unsigned tile_h,
                                                  unsigned tensor_w, unsigned tensor_h,
                                                  unsigned stride) {
    unsigned long long ga = (unsigned long long)gaddr;
    u32x4 g0;
    g0[0] = 1u;                                            // count=1, user desc
    g0[1] = lds_off;                                       // lds_addr (bytes)
    g0[2] = (unsigned)(ga & 0xffffffffu);                  // global_addr[31:0]
    g0[3] = (unsigned)((ga >> 32) & 0x01ffffffu) | (2u << 30);  // addr[56:32] | type=2
    i32x8 g1;
    g1[0] = (int)(2u << 16);                               // wg_mask=0, data_size=4B
    g1[1] = (int)((tensor_w & 0xffffu) << 16);             // tensor_dim0[15:0]
    g1[2] = (int)(((tensor_h & 0xffffu) << 16) | ((tensor_w >> 16) & 0xffffu));
    g1[3] = (int)(((tile_w & 0xffffu) << 16) | ((tensor_h >> 16) & 0xffffu));
    g1[4] = (int)(tile_h & 0xffffu);                       // tile_dim1; tile_dim2=0
    g1[5] = (int)stride;                                   // dim0_stride[31:0]
    g1[6] = 0;                                             // dim0_stride[47:32], dim1_stride lo
    g1[7] = 0;
    i32x4 gz = {0, 0, 0, 0};                               // 2D: groups 2/3 unused
#if defined(__clang_major__) && (__clang_major__ >= 23)
    i32x8 z8 = {0, 0, 0, 0, 0, 0, 0, 0};
    __builtin_amdgcn_tensor_load_to_lds(g0, g1, gz, gz, z8, 0);
#else
    __builtin_amdgcn_tensor_load_to_lds(g0, g1, gz, gz, 0);
#endif
}
#endif

// ---------- fp32 -> bf16 conversion ----------
__global__ void cvt_bf16_kernel(const float* __restrict__ in,
                                unsigned short* __restrict__ out, int n) {
    int idx = blockIdx.x * 256 + threadIdx.x;
    if (idx < n) out[idx] = f2bf(in[idx]);
}

// ---------- Q/K/V/skip projections: 4x [2048x512] = [2048x512]@[512x512]^T ----------
__global__ void proj_gemm_kernel(const unsigned short* __restrict__ xbf,
                                 const unsigned short* __restrict__ w0,
                                 const unsigned short* __restrict__ w1,
                                 const unsigned short* __restrict__ w2,
                                 const unsigned short* __restrict__ w3,
                                 float* __restrict__ o0, float* __restrict__ o1,
                                 float* __restrict__ o2, float* __restrict__ o3) {
    int tid  = threadIdx.x;
    int lane = tid & 31;
    int wv   = tid >> 5;
    int gw   = blockIdx.x * 8 + wv;          // 0..16383
    int mi   = gw >> 12;                     // matrix id 0..3
    int rem  = gw & 4095;
    int mb   = rem >> 5;                     // 0..127
    int nb   = rem & 31;                     // 0..31
    const unsigned short* W = (mi == 0) ? w0 : (mi == 1) ? w1 : (mi == 2) ? w2 : w3;
    float* O = (mi == 0) ? o0 : (mi == 1) ? o1 : (mi == 2) ? o2 : o3;

    v8f acc = {};
    for (int kt = 0; kt < 16; ++kt) {
        v16bf a  = load_bf16_frag(xbf + (mb * 16) * EMB + kt * 32, EMB, lane);
        v16bf bb = load_bf16_frag(W   + (nb * 16) * EMB + kt * 32, EMB, lane);
        acc = __builtin_amdgcn_wmma_f32_16x16x32_bf16(false, a, false, bb,
                                                      (short)0, acc, false, false);
    }
    int col = lane & 15;
    int rb  = (lane >> 4) << 3;
    for (int r = 0; r < 8; ++r)
        O[(size_t)(mb * 16 + rb + r) * EMB + nb * 16 + col] = acc[r];
}

// ---------- fused knowledge attention ----------
// grid = BSZ*SLEN blocks (one per (b, query i)), 256 threads (8 waves).
// LDS layout (bytes):
//   stage0 f32[128*32]  @ 0        (16384)   TDM staging, double buffered
//   stage1 f32[128*32]  @ 16384    (16384)
//   knowBF bf16[128*512]@ 32768    (131072)
//   knBF   bf16[128*512]@ 163840   (131072)
//   qrow   f32[512]     @ 294912   (2048)
//   sc     f32[8*128]   @ 296960   (4096)
#define CW          32                       // K-chunk width (columns)
#define NCHUNK      (EMB / CW)               // 16
#define STAGE_BYTES (128 * CW * 4)           // 16384
#define SMEM_BYTES  (2 * STAGE_BYTES + 2 * 128 * EMB * 2 + EMB * 4 + NHEAD * 128 * 4)

__global__ void attn_kernel(const float* __restrict__ know,
                            const float* __restrict__ qlin,
                            const float* __restrict__ klin,
                            const float* __restrict__ vlin,
                            const unsigned short* __restrict__ WknowBF,
                            const int* __restrict__ mask,
                            float* __restrict__ attn) {
    extern __shared__ char smem_raw[];
    float*          stage0 = (float*)smem_raw;
    float*          stage1 = (float*)(smem_raw + STAGE_BYTES);
    unsigned short* knowBF = (unsigned short*)(smem_raw + 2 * STAGE_BYTES);
    unsigned short* knBF   = knowBF + 128 * EMB;
    float* qrow = (float*)(knBF + 128 * EMB);
    float* sc   = qrow + EMB;

    int tid  = threadIdx.x;
    int lane = tid & 31;
    int wv   = tid >> 5;
    int b = blockIdx.x & (BSZ - 1);
    int i = blockIdx.x >> 4;

    const float* tile_base =
        know + ((size_t)b * (SLEN * SLEN) + (size_t)i * SLEN) * EMB;

    {   // q row for (i, b): all 8 heads concatenated
        const float* q = qlin + ((size_t)i * BSZ + b) * EMB;
        qrow[tid] = q[tid];
        qrow[tid + 256] = q[tid + 256];
    }

#if HAVE_TDM
    // ---- Phase A: TDM-pipelined stream of the know tile, f32 -> bf16 in LDS ----
    unsigned stage_off[2];
    stage_off[0] = (unsigned)(size_t)stage0;   // low 32 bits of LDS generic addr
    stage_off[1] = (unsigned)(size_t)stage1;
    if (wv == 0) {
        tdm_load_tile_f32(tile_base, stage_off[0], CW, 128, EMB, 128, EMB);
    }
    for (int c = 0; c < NCHUNK; ++c) {
        if (wv == 0) __builtin_amdgcn_s_wait_tensorcnt((short)0);  // chunk c landed
        __syncthreads();
        if (wv == 0 && (c + 1) < NCHUNK) {                         // overlap next DMA
            tdm_load_tile_f32(tile_base + (c + 1) * CW, stage_off[(c + 1) & 1],
                              CW, 128, EMB, 128, EMB);
        }
        const float4* s4 = (const float4*)((c & 1) ? stage1 : stage0);
        for (int r = 0; r < 4; ++r) {                              // convert chunk c
            int fidx = tid + 256 * r;        // 1024 float4 per chunk
            int j = fidx >> 3;               // 8 float4 per 32-col row
            int g = fidx & 7;
            float4 f = s4[fidx];
            int o = j * EMB + c * CW + g * 4;
            knowBF[o + 0] = f2bf(f.x); knowBF[o + 1] = f2bf(f.y);
            knowBF[o + 2] = f2bf(f.z); knowBF[o + 3] = f2bf(f.w);
        }
        __syncthreads();                                           // stage reusable
    }
#else
    // ---- Phase A (fallback): synchronous load + convert ----
    const float4* src = (const float4*)tile_base;
    for (int t = tid; t < 128 * EMB / 4; t += 256) {
        float4 f = src[t];
        int o = t * 4;
        knowBF[o + 0] = f2bf(f.x); knowBF[o + 1] = f2bf(f.y);
        knowBF[o + 2] = f2bf(f.z); knowBF[o + 3] = f2bf(f.w);
    }
    __syncthreads();
#endif

    // ---- Phase B: knBF(128x512) = knowBF(128x512) @ Wknow^T via bf16 WMMA ----
    int mb = wv;                                  // each wave owns 16 rows of j
    for (int nb = 0; nb < 32; ++nb) {
        v8f acc = {};
        for (int kt = 0; kt < 16; ++kt) {
            v16bf a  = load_bf16_frag(knowBF + (mb * 16) * EMB + kt * 32, EMB, lane);
            v16bf bb = load_bf16_frag(WknowBF + (nb * 16) * EMB + kt * 32, EMB, lane);
            acc = __builtin_amdgcn_wmma_f32_16x16x32_bf16(false, a, false, bb,
                                                          (short)0, acc, false, false);
        }
        int col = lane & 15;
        int rb  = (lane >> 4) << 3;
        for (int r = 0; r < 8; ++r)
            knBF[(mb * 16 + rb + r) * EMB + nb * 16 + col] = f2bf(acc[r]);
    }
    __syncthreads();

    // ---- Phase C: scores  s[h][j] = q_i . (k_j + kn_ij), mask, scale ----
    const float scale = 0.125f;                   // D^-0.5, D=64
    int j   = tid & 127;
    int hh0 = tid >> 7;                           // 0 or 1
    for (int hp = 0; hp < 4; ++hp) {
        int h = hp * 2 + hh0;
        const float*          krow = klin + ((size_t)j * BSZ + b) * EMB + h * HEAD_D;
        const unsigned short* knr  = knBF + j * EMB + h * HEAD_D;
        const float*          qh   = qrow + h * HEAD_D;
        float s = 0.f;
        for (int d = 0; d < HEAD_D; ++d)
            s += qh[d] * (krow[d] + bf2f(knr[d]));
        int m = mask[((size_t)(b * NHEAD + h)) * (SLEN * SLEN) + (size_t)i * SLEN + j];
        sc[h * 128 + j] = m ? s * scale : NEG_INF;
    }
    __syncthreads();

    // ---- Phase D: softmax over j; wave w handles head w (wave32 shfl) ----
    {
        int h = wv;
        float vals[4];
        float mx = -3.4e38f;
        for (int r = 0; r < 4; ++r) {
            vals[r] = sc[h * 128 + lane + r * 32];
            mx = fmaxf(mx, vals[r]);
        }
        for (int o = 16; o > 0; o >>= 1) mx = fmaxf(mx, __shfl_xor(mx, o, 32));
        float sum = 0.f;
        for (int r = 0; r < 4; ++r) { vals[r] = __expf(vals[r] - mx); sum += vals[r]; }
        for (int o = 16; o > 0; o >>= 1) sum += __shfl_xor(sum, o, 32);
        float inv = 1.0f / sum;
        for (int r = 0; r < 4; ++r) sc[h * 128 + lane + r * 32] = vals[r] * inv;
    }
    __syncthreads();

    // ---- Phase E: out[h][d] = sum_j p[j] * (v[j,d] + kn[j,d]) ----
    for (int rr = 0; rr < 2; ++rr) {
        int idx = tid + 256 * rr;                 // 512 outputs = 8 heads x 64 d
        int h = idx >> 6, d = idx & 63;
        const float* p = sc + h * 128;
        float acc = 0.f;
        for (int jj = 0; jj < 128; ++jj)
            acc += p[jj] * (vlin[((size_t)jj * BSZ + b) * EMB + h * HEAD_D + d] +
                            bf2f(knBF[jj * EMB + h * HEAD_D + d]));
        attn[((size_t)i * BSZ + b) * EMB + h * HEAD_D + d] = acc;
    }
}

// ---------- gating ----------
__global__ void gate_kernel(const float* __restrict__ attn,
                            const float* __restrict__ rlin,
                            const float* __restrict__ Wbeta,
                            float* __restrict__ out) {
    __shared__ float red[8];
    __shared__ float gsh;
    int row = blockIdx.x;
    int tid = threadIdx.x;
    const float* a  = attn + (size_t)row * EMB;
    const float* rv = rlin + (size_t)row * EMB;
    float part = 0.f;
    for (int e = tid; e < EMB; e += 256) {
        float av = a[e], rr = rv[e];
        part += av * Wbeta[e] + rr * Wbeta[EMB + e] + (av - rr) * Wbeta[2 * EMB + e];
    }
    int lane = tid & 31, wv = tid >> 5;
    for (int o = 16; o > 0; o >>= 1) part += __shfl_xor(part, o, 32);
    if (lane == 0) red[wv] = part;
    __syncthreads();
    if (tid == 0) {
        float t = 0.f;
        for (int k = 0; k < 8; ++k) t += red[k];
        gsh = 1.0f / (1.0f + __expf(-t));
    }
    __syncthreads();
    float g = gsh;
    for (int e = tid; e < EMB; e += 256) {
        float av = a[e], rr = rv[e];
        out[(size_t)row * EMB + e] = g * rr + (1.f - g) * av;
    }
}

// ---------- host side ----------
extern "C" void kernel_launch(void* const* d_in, const int* in_sizes, int n_in,
                              void* d_out, int out_size, void* d_ws, size_t ws_size,
                              hipStream_t stream) {
    (void)in_sizes; (void)n_in; (void)out_size; (void)ws_size;
    const float* x     = (const float*)d_in[0];
    const float* know  = (const float*)d_in[1];
    const int*   mask  = (const int*)d_in[2];
    const float* Wq    = (const float*)d_in[3];
    const float* Wk    = (const float*)d_in[4];
    const float* Wv    = (const float*)d_in[5];
    const float* Wknow = (const float*)d_in[6];
    const float* Wskip = (const float*)d_in[7];
    const float* Wbeta = (const float*)d_in[8];

    char* ws = (char*)d_ws;
    const size_t X_ELEMS = (size_t)ROWS * EMB;    // 1,048,576
    const size_t W_ELEMS = (size_t)EMB * EMB;     // 262,144
    unsigned short* xBF     = (unsigned short*)ws;             ws += X_ELEMS * 2;
    unsigned short* WqBF    = (unsigned short*)ws;             ws += W_ELEMS * 2;
    unsigned short* WkBF    = (unsigned short*)ws;             ws += W_ELEMS * 2;
    unsigned short* WvBF    = (unsigned short*)ws;             ws += W_ELEMS * 2;
    unsigned short* WskipBF = (unsigned short*)ws;             ws += W_ELEMS * 2;
    unsigned short* WknowBF = (unsigned short*)ws;             ws += W_ELEMS * 2;
    float* qlin    = (float*)ws;                               ws += X_ELEMS * 4;
    float* klin    = (float*)ws;                               ws += X_ELEMS * 4;
    float* vlin    = (float*)ws;                               ws += X_ELEMS * 4;
    float* rlin    = (float*)ws;                               ws += X_ELEMS * 4;
    float* attnbuf = (float*)ws;                               ws += X_ELEMS * 4;

    cvt_bf16_kernel<<<(int)((X_ELEMS + 255) / 256), 256, 0, stream>>>(x, xBF, (int)X_ELEMS);
    cvt_bf16_kernel<<<(int)((W_ELEMS + 255) / 256), 256, 0, stream>>>(Wq, WqBF, (int)W_ELEMS);
    cvt_bf16_kernel<<<(int)((W_ELEMS + 255) / 256), 256, 0, stream>>>(Wk, WkBF, (int)W_ELEMS);
    cvt_bf16_kernel<<<(int)((W_ELEMS + 255) / 256), 256, 0, stream>>>(Wv, WvBF, (int)W_ELEMS);
    cvt_bf16_kernel<<<(int)((W_ELEMS + 255) / 256), 256, 0, stream>>>(Wskip, WskipBF, (int)W_ELEMS);
    cvt_bf16_kernel<<<(int)((W_ELEMS + 255) / 256), 256, 0, stream>>>(Wknow, WknowBF, (int)W_ELEMS);

    proj_gemm_kernel<<<2048, 256, 0, stream>>>(xBF, WqBF, WkBF, WvBF, WskipBF,
                                               qlin, klin, vlin, rlin);

    attn_kernel<<<BSZ * SLEN, 256, SMEM_BYTES, stream>>>(know, qlin, klin, vlin,
                                                         WknowBF, mask, attnbuf);

    gate_kernel<<<ROWS, 256, 0, stream>>>(attnbuf, rlin, Wbeta, (float*)d_out);
}
